// BigramLanguageModel_42331197669544
// MI455X (gfx1250) — matrix operand order repaired and checked by reading, not compiled, not run on hardware
//
#include <hip/hip_runtime.h>
#include <hip/hip_bf16.h>
#include <math.h>

typedef __attribute__((ext_vector_type(16))) __bf16 v16bf;
typedef __attribute__((ext_vector_type(4)))  __bf16 v4bf;
typedef __attribute__((ext_vector_type(2)))  __bf16 v2bf;
typedef __attribute__((ext_vector_type(8)))  float  v8f;
typedef __attribute__((ext_vector_type(4)))  float  v4f;
typedef __attribute__((ext_vector_type(4)))  unsigned int u32x4;

// Problem constants (fixed by the reference)
static constexpr int B_  = 64;
static constexpr int T_  = 256;
static constexpr int C_  = 256;
static constexpr int V_  = 65;
static constexpr int L_  = 6;
static constexpr int H_  = 8;
static constexpr int HS_ = 32;
static constexpr int BT_ = B_ * T_;          // 16384
static constexpr float SCALE_ = 0.0625f;     // C^-0.5 = 1/16

// ---------------------------------------------------------------------------
// Embedding: x[b,t,c] = tok_emb[idx[b,t], c] + pos_emb[t, c]
// ---------------------------------------------------------------------------
__global__ __launch_bounds__(256) void embed_kernel(const int* __restrict__ idx,
                                                    const float* __restrict__ tok,
                                                    const float* __restrict__ pos,
                                                    float* __restrict__ x) {
  int i  = blockIdx.x * 256 + threadIdx.x;   // < BT*C
  int bt = i >> 8;
  int c  = i & 255;
  int t  = bt & 255;
  x[i] = tok[idx[bt] * C_ + c] + pos[t * C_ + c];
}

// ---------------------------------------------------------------------------
// LayerNorm: one wave32 per row of 256, 8 elements per lane, shuffle reduce
// ---------------------------------------------------------------------------
__global__ __launch_bounds__(256) void ln_kernel(const float* __restrict__ x,
                                                 const float* __restrict__ g,
                                                 const float* __restrict__ b,
                                                 float* __restrict__ y, int rows) {
  int row  = (blockIdx.x * blockDim.x + threadIdx.x) >> 5;
  int lane = threadIdx.x & 31;
  if (row >= rows) return;
  const float* xr = x + (size_t)row * C_;
  float v[8];
  float s = 0.f;
#pragma unroll
  for (int i = 0; i < 8; ++i) { v[i] = xr[lane + i * 32]; s += v[i]; }
#pragma unroll
  for (int off = 16; off > 0; off >>= 1) s += __shfl_xor(s, off, 32);
  float mu = s * (1.f / C_);
  float t2 = 0.f;
#pragma unroll
  for (int i = 0; i < 8; ++i) { float d = v[i] - mu; t2 += d * d; }
#pragma unroll
  for (int off = 16; off > 0; off >>= 1) t2 += __shfl_xor(t2, off, 32);
  float rstd = rsqrtf(t2 * (1.f / C_) + 1e-5f);
  float* yr = y + (size_t)row * C_;
#pragma unroll
  for (int i = 0; i < 8; ++i) {
    int c = lane + i * 32;
    yr[c] = (v[i] - mu) * rstd * g[c] + b[c];
  }
}

// ---------------------------------------------------------------------------
// WMMA GEMM: out[M,N] (+epilogue) = A[M,K] * B[K,N]
//   Block tile 128x128, 256 threads (8 waves as 4x2), wave tile 32x64:
//   2 A-fragments x 4 B-fragments -> 8 WMMAs per 32-wide K step.
//   LDS: A row-major [128][48] bf16 (96B rows); B transposed [n][k] [128][48]
//   so every WMMA fragment is exactly 2x 16-byte LDS loads.
//   AMODE: 0 = row-major [M,K]; 1 = read head-concat from [B,H,T,HS] scratch
//   BMODE: 0 = row-major [K,N]; 1 = qkv weight layout [H, C(K), HS]
//   EPI bit0 = +bias[n], bit1 = ReLU, bit2 = +res[m,n] (row-major, ld=N)
//   SMODE: 0 = row-major out [M,N]; 1 = scatter to [B,H,T,HS]
// ---------------------------------------------------------------------------
template <int AMODE, int BMODE, int EPI, int SMODE>
__global__ __launch_bounds__(256) void gemm_wmma_kernel(
    const float* __restrict__ A, const float* __restrict__ Bm,
    const float* __restrict__ bias, const float* __restrict__ res,
    float* __restrict__ out, int M, int N, int K) {
  constexpr int BM = 128, BN = 128, BK = 32;
  constexpr int LDA = BK + 16;   // 48 bf16 = 96 bytes (16B aligned rows)
  constexpr int LDB = BK + 16;   // transposed B rows: 48 bf16
  __shared__ __align__(16) __bf16 As[BM][LDA];
  __shared__ __align__(16) __bf16 Bt[BN][LDB];

  const int tid  = threadIdx.x;
  const int lane = tid & 31;
  const int w    = tid >> 5;
  const int wm   = (w & 3) * 32;   // wave row offset in block tile
  const int wn   = (w >> 2) * 64;  // wave col offset in block tile
  const int m0   = blockIdx.x * BM;
  const int n0   = blockIdx.y * BN;

  v8f acc[8];
#pragma unroll
  for (int t = 0; t < 8; ++t) acc[t] = v8f{};

  const bool nvec = ((N & 3) == 0);  // vector-safe B/global path

  for (int k0 = 0; k0 < K; k0 += BK) {
    // --- stage A tile (128x32) : 4 float4 chunks per thread ---
#pragma unroll
    for (int i = 0; i < 4; ++i) {
      int chunk = i * 256 + tid;          // < 1024
      int r = chunk >> 3;                 // 0..127
      int c = (chunk & 7) * 4;            // 0..28
      int mm = m0 + r, kk = k0 + c;
      v4f va;
      if constexpr (AMODE == 0) {
        va = *(const v4f*)(A + (size_t)mm * K + kk);
      } else {  // [B,H,T,HS] gather: contiguous in d since c is 4-aligned, BK=32
        int bb = mm >> 8, tt = mm & 255, hh = kk >> 5, dd = kk & 31;
        va = *(const v4f*)(A + ((((size_t)bb << 3) + hh) * 256 + tt) * 32 + dd);
      }
      v4bf pk = {(__bf16)va.x, (__bf16)va.y, (__bf16)va.z, (__bf16)va.w};
      *(v4bf*)(&As[r][c]) = pk;
    }
    // --- stage B tile (32x128) transposed : 2 chunks of 2 rows x 4 cols ---
#pragma unroll
    for (int i = 0; i < 2; ++i) {
      int chunk = i * 256 + tid;          // < 512
      int kp = chunk >> 5;                // 0..15 -> rows 2kp, 2kp+1
      int n  = (chunk & 31) * 4;          // 0..124
      int kk = k0 + 2 * kp;
      int nn = n0 + n;
      v4f r0, r1;
      if constexpr (BMODE == 0) {
        if (nvec && nn < N) {
          r0 = *(const v4f*)(Bm + (size_t)kk * N + nn);
          r1 = *(const v4f*)(Bm + (size_t)(kk + 1) * N + nn);
        } else {
#pragma unroll
          for (int j = 0; j < 4; ++j) {
            r0[j] = (nn + j < N) ? Bm[(size_t)kk * N + nn + j] : 0.f;
            r1[j] = (nn + j < N) ? Bm[(size_t)(kk + 1) * N + nn + j] : 0.f;
          }
        }
      } else {  // weight [H, K, HS]: n 4-aligned, never crosses head boundary
        int hh = nn >> 5, dd = nn & 31;
        r0 = *(const v4f*)(Bm + (((size_t)hh * K + kk) * 32) + dd);
        r1 = *(const v4f*)(Bm + (((size_t)hh * K + kk + 1) * 32) + dd);
      }
#pragma unroll
      for (int j = 0; j < 4; ++j) {
        v2bf pk = {(__bf16)r0[j], (__bf16)r1[j]};
        *(v2bf*)(&Bt[n + j][2 * kp]) = pk;
      }
    }
    // prefetch next K tile into L2 while we compute
    if (k0 + BK < K) {
      if constexpr (AMODE == 0)
        __builtin_prefetch(A + (size_t)(m0 + (tid >> 1)) * K + k0 + BK, 0, 1);
      if constexpr (BMODE == 0)
        __builtin_prefetch(Bm + (size_t)(k0 + BK + (tid & 31)) * N + n0, 0, 1);
    }
    __syncthreads();

    // --- fragments: each is two 16B LDS loads ---
    const int kaIdx = (lane & 16) ? 1 : 0;  // A: K halves {0..7 / 16..23} vs {8..15 / 24..31}
    v16bf af[2];
#pragma unroll
    for (int mi = 0; mi < 2; ++mi) {
      int mr = wm + mi * 16 + (lane & 15);
      const u32x4* rowp = (const u32x4*)(&As[mr][0]);
      union { u32x4 q[2]; v16bf v; } ua;
      ua.q[0] = rowp[kaIdx];
      ua.q[1] = rowp[2 + kaIdx];
      af[mi] = ua.v;
    }
    const int kbIdx = (lane & 16) ? 1 : 0;  // B: K halves {0..7 / 16..23} vs {8..15 / 24..31}
#pragma unroll
    for (int ni = 0; ni < 4; ++ni) {
      int nc = wn + ni * 16 + (lane & 15);
      const u32x4* rowp = (const u32x4*)(&Bt[nc][0]);
      union { u32x4 q[2]; v16bf v; } ub;
      ub.q[0] = rowp[kbIdx];
      ub.q[1] = rowp[2 + kbIdx];
      v16bf bfrag = ub.v;
#pragma unroll
      for (int mi = 0; mi < 2; ++mi)
        acc[mi * 4 + ni] = __builtin_amdgcn_wmma_f32_16x16x32_bf16(
            false, af[mi], false, bfrag, (short)0, acc[mi * 4 + ni], false, false);
    }
    __syncthreads();
  }

  // --- epilogue (D layout: VGPR r -> m = r + 8*(lane>=16), n = lane&15) ---
#pragma unroll
  for (int mi = 0; mi < 2; ++mi) {
    const int mbase = m0 + wm + mi * 16 + ((lane & 16) >> 1);
#pragma unroll
    for (int ni = 0; ni < 4; ++ni) {
      const int n = n0 + wn + ni * 16 + (lane & 15);
      if (n < N) {
#pragma unroll
        for (int r = 0; r < 8; ++r) {
          int m = mbase + r;
          float v = acc[mi * 4 + ni][r];
          if constexpr ((EPI & 1) != 0) v += bias[n];
          if constexpr ((EPI & 2) != 0) v = fmaxf(v, 0.f);
          if constexpr ((EPI & 4) != 0) v += res[(size_t)m * N + n];
          if constexpr (SMODE == 0) {
            out[(size_t)m * N + n] = v;
          } else {  // scatter to [B,H,T,HS]
            int bb = m >> 8, tt = m & 255, hh = n >> 5, dd = n & 31;
            out[((((size_t)bb << 3) + hh) * 256 + tt) * 32 + dd] = v;
          }
        }
      }
    }
  }
}

// ---------------------------------------------------------------------------
// Attention: one wave per (b,h, 16-row query tile).
//   S = Q*K^T via WMMA (K-dim = HS = 32 -> single WMMA per 16x16 tile),
//   causal mask + softmax in LDS (fp32), O = P*V via WMMA (P -> bf16).
// ---------------------------------------------------------------------------
__global__ __launch_bounds__(32) void attn_kernel(const float* __restrict__ q,
                                                  const float* __restrict__ k,
                                                  const float* __restrict__ v,
                                                  float* __restrict__ o) {
  __shared__ float sP[16][T_ + 4];
  const int lane = threadIdx.x;
  const int mt   = blockIdx.x & 15;     // query row tile
  const int bh   = blockIdx.x >> 4;     // b*H + h

  // Q fragment: rows mt*16..mt*16+15, K = HS = 32
  const int mrow = mt * 16 + (lane & 15);
  const int ka   = (lane & 16) ? 8 : 0;
  const float* qrow = q + ((size_t)bh * T_ + mrow) * HS_;
  v16bf qa;
#pragma unroll
  for (int i = 0; i < 8; ++i) {
    qa[i]     = (__bf16)qrow[ka + i];
    qa[8 + i] = (__bf16)qrow[ka + 16 + i];
  }

  // Scores: 16 key tiles of 16
  for (int nt = 0; nt < 16; ++nt) {
    const int ncol = nt * 16 + (lane & 15);      // key index
    const int kb   = (lane & 16) ? 8 : 0;
    const float* krow = k + ((size_t)bh * T_ + ncol) * HS_ + kb;
    v16bf kf;
#pragma unroll
    for (int j = 0; j < 16; ++j) kf[j] = (__bf16)krow[(j < 8) ? j : j + 8];
    v8f s = v8f{};
    s = __builtin_amdgcn_wmma_f32_16x16x32_bf16(false, qa, false, kf,
                                                (short)0, s, false, false);
#pragma unroll
    for (int r = 0; r < 8; ++r) {
      int ml = r + ((lane & 16) >> 1);
      int tq = mt * 16 + ml;
      int tk = nt * 16 + (lane & 15);
      sP[ml][tk] = (tk <= tq) ? s[r] * SCALE_ : -__builtin_inff();
    }
  }
  __syncthreads();

  // Softmax over 256 per row (32 lanes cooperate, 8 elems each)
  for (int r = 0; r < 16; ++r) {
    float mx = -__builtin_inff();
    for (int i = lane; i < T_; i += 32) mx = fmaxf(mx, sP[r][i]);
#pragma unroll
    for (int off = 16; off > 0; off >>= 1) mx = fmaxf(mx, __shfl_xor(mx, off, 32));
    float sum = 0.f;
    for (int i = lane; i < T_; i += 32) {
      float e = __expf(sP[r][i] - mx);
      sP[r][i] = e;
      sum += e;
    }
#pragma unroll
    for (int off = 16; off > 0; off >>= 1) sum += __shfl_xor(sum, off, 32);
    float inv = 1.f / sum;
    for (int i = lane; i < T_; i += 32) sP[r][i] *= inv;
  }
  __syncthreads();

  // O = P * V : K loop over 256 in steps of 32; two 16-col output tiles
  v8f acc[2] = {v8f{}, v8f{}};
  for (int k0 = 0; k0 < T_; k0 += 32) {
    const int m  = lane & 15;
    const int kp = (lane & 16) ? 8 : 0;
    v16bf pa;
#pragma unroll
    for (int i = 0; i < 8; ++i) {
      pa[i]     = (__bf16)sP[m][k0 + kp + i];
      pa[8 + i] = (__bf16)sP[m][k0 + kp + 16 + i];
    }
#pragma unroll
    for (int nt = 0; nt < 2; ++nt) {
      const int nc = nt * 16 + (lane & 15);
      const int kb = (lane & 16) ? 8 : 0;
      v16bf vf;
#pragma unroll
      for (int j = 0; j < 16; ++j)
        vf[j] = (__bf16)v[((size_t)bh * T_ + k0 + kb + ((j < 8) ? j : j + 8)) * HS_ + nc];
      acc[nt] = __builtin_amdgcn_wmma_f32_16x16x32_bf16(
          false, pa, false, vf, (short)0, acc[nt], false, false);
    }
  }
#pragma unroll
  for (int nt = 0; nt < 2; ++nt)
#pragma unroll
    for (int r = 0; r < 8; ++r) {
      int ml = r + ((lane & 16) >> 1);
      o[((size_t)bh * T_ + mt * 16 + ml) * HS_ + nt * 16 + (lane & 15)] = acc[nt][r];
    }
}

// ---------------------------------------------------------------------------
// Loss: per-row log-sum-exp over 65 logits, one atomic per row
// ---------------------------------------------------------------------------
__global__ void zero_loss_kernel(float* p) { *p = 0.f; }

__global__ __launch_bounds__(32) void loss_kernel(const float* __restrict__ logits,
                                                  const int* __restrict__ targets,
                                                  float* __restrict__ loss) {
  int row  = blockIdx.x;
  int lane = threadIdx.x;
  const float* lr = logits + (size_t)row * V_;
  float mx = -__builtin_inff();
  for (int i = lane; i < V_; i += 32) mx = fmaxf(mx, lr[i]);
#pragma unroll
  for (int off = 16; off > 0; off >>= 1) mx = fmaxf(mx, __shfl_xor(mx, off, 32));
  float s = 0.f;
  for (int i = lane; i < V_; i += 32) s += __expf(lr[i] - mx);
#pragma unroll
  for (int off = 16; off > 0; off >>= 1) s += __shfl_xor(s, off, 32);
  if (lane == 0) {
    float lse = mx + __logf(s);
    atomicAdd(loss, (lse - lr[targets[row]]) * (1.f / (float)BT_));
  }
}

// ---------------------------------------------------------------------------
// Host orchestration
// ---------------------------------------------------------------------------
extern "C" void kernel_launch(void* const* d_in, const int* in_sizes, int n_in,
                              void* d_out, int out_size, void* d_ws, size_t ws_size,
                              hipStream_t stream) {
  (void)in_sizes; (void)n_in; (void)out_size; (void)ws_size;

  const int*   idx     = (const int*)d_in[0];
  const int*   targets = (const int*)d_in[1];
  const float* tok_emb = (const float*)d_in[2];
  const float* pos_emb = (const float*)d_in[3];
  const float* wq   = (const float*)d_in[4];
  const float* bq   = (const float*)d_in[5];
  const float* wk   = (const float*)d_in[6];
  const float* bk   = (const float*)d_in[7];
  const float* wv   = (const float*)d_in[8];
  const float* bv   = (const float*)d_in[9];
  const float* wo   = (const float*)d_in[10];
  const float* bo   = (const float*)d_in[11];
  const float* ln1g = (const float*)d_in[12];
  const float* ln1b = (const float*)d_in[13];
  const float* w1   = (const float*)d_in[14];
  const float* b1   = (const float*)d_in[15];
  const float* w2   = (const float*)d_in[16];
  const float* b2   = (const float*)d_in[17];
  const float* ln2g = (const float*)d_in[18];
  const float* ln2b = (const float*)d_in[19];
  const float* lnfg = (const float*)d_in[20];
  const float* lnfb = (const float*)d_in[21];
  const float* lm_w = (const float*)d_in[22];
  const float* lm_b = (const float*)d_in[23];

  // workspace carving (all fp32)
  float* x  = (float*)d_ws;                       // [BT, C]
  float* xn = x  + (size_t)BT_ * C_;              // [BT, C]
  float* qs = xn + (size_t)BT_ * C_;              // [B,H,T,HS]
  float* ks = qs + (size_t)BT_ * C_;
  float* vs = ks + (size_t)BT_ * C_;
  float* os = vs + (size_t)BT_ * C_;
  float* hs = os + (size_t)BT_ * C_;              // [BT, 4C]

  float* logits = (float*)d_out;                  // [BT, V]
  float* loss   = logits + (size_t)BT_ * V_;      // scalar

  const size_t wstrideQ = (size_t)H_ * C_ * HS_;  // 65536 per layer
  const size_t wstrideM = (size_t)C_ * 4 * C_;    // 262144 per layer

  dim3 blk256(256);
  dim3 gC(BT_ / 128, C_ / 128);         // N = 256 -> y = 2
  dim3 gF(BT_ / 128, (4 * C_) / 128);   // N = 1024 -> y = 8
  dim3 gV(BT_ / 128, 1);                // N = 65 -> y = 1

  embed_kernel<<<(BT_ * C_) / 256, blk256, 0, stream>>>(idx, tok_emb, pos_emb, x);

  for (int l = 0; l < L_; ++l) {
    ln_kernel<<<BT_ / 8, blk256, 0, stream>>>(x, ln1g + l * C_, ln1b + l * C_, xn, BT_);

    // QKV: A row-major, B in [H,C,HS] layout, +bias, scatter to [B,H,T,HS]
    gemm_wmma_kernel<0, 1, 1, 1><<<gC, blk256, 0, stream>>>(
        xn, wq + l * wstrideQ, bq + l * C_, nullptr, qs, BT_, C_, C_);
    gemm_wmma_kernel<0, 1, 1, 1><<<gC, blk256, 0, stream>>>(
        xn, wk + l * wstrideQ, bk + l * C_, nullptr, ks, BT_, C_, C_);
    gemm_wmma_kernel<0, 1, 1, 1><<<gC, blk256, 0, stream>>>(
        xn, wv + l * wstrideQ, bv + l * C_, nullptr, vs, BT_, C_, C_);

    attn_kernel<<<B_ * H_ * (T_ / 16), dim3(32), 0, stream>>>(qs, ks, vs, os);

    // out proj: A = head-concat gather, +bias +residual, in-place into x
    gemm_wmma_kernel<1, 0, 5, 0><<<gC, blk256, 0, stream>>>(
        os, wo + l * (size_t)C_ * C_, bo + l * C_, x, x, BT_, C_, C_);

    ln_kernel<<<BT_ / 8, blk256, 0, stream>>>(x, ln2g + l * C_, ln2b + l * C_, xn, BT_);

    // MLP: fc1 (+bias, ReLU), fc2 (+bias, +residual)
    gemm_wmma_kernel<0, 0, 3, 0><<<gF, blk256, 0, stream>>>(
        xn, w1 + l * wstrideM, b1 + l * 4 * C_, nullptr, hs, BT_, 4 * C_, C_);
    gemm_wmma_kernel<0, 0, 5, 0><<<gC, blk256, 0, stream>>>(
        hs, w2 + l * wstrideM, b2 + l * C_, x, x, BT_, C_, 4 * C_);
  }

  ln_kernel<<<BT_ / 8, blk256, 0, stream>>>(x, lnfg, lnfb, xn, BT_);

  // lm_head: logits straight into d_out (N=65 guarded, scalar B path)
  gemm_wmma_kernel<0, 0, 1, 0><<<gV, blk256, 0, stream>>>(
      xn, lm_w, lm_b, nullptr, logits, BT_, V_, C_);

  zero_loss_kernel<<<1, 1, 0, stream>>>(loss);
  loss_kernel<<<BT_, dim3(32), 0, stream>>>(logits, targets, loss);
}
